// MultiHeadLocalAttention_46737834115410
// MI455X (gfx1250) — matrix operand order repaired and checked by reading, not compiled, not run on hardware
//
#include <hip/hip_runtime.h>
#include <math.h>

// MultiHeadLocalAttention for MI455X (gfx1250, wave32, WMMA)
// N=32768, C=128, H=4, D=32, K=16
#define NPTS 32768
#define CDIM 128
#define HEADS 4
#define DHEAD 32
#define KNN 16

typedef __attribute__((ext_vector_type(16))) _Float16 v16h;
typedef __attribute__((ext_vector_type(8)))  float    v8f;

// ---------------------------------------------------------------------------
// WMMA fragment loaders (fp32 source, f16 fragments), contiguous k order, same for A and B.
// A: 16x32 (MxK). lane<16 -> M=lane, elems 0..15 = K 0..15
//    lane>=16 -> M=lane-16, elems 0..15 = K 16..31
// B: 32x16 (KxN). lane -> N = lane%16; lane<16 holds K 0..15, lane>=16 K 16..31
// ---------------------------------------------------------------------------
__device__ inline v16h load_a_frag(const float* __restrict__ X, int m0, int kc, int lane) {
    const int half = lane >> 4;
    const int m    = lane & 15;
    const float* row = X + (size_t)(m0 + m) * CDIM + kc * 32 + half * 16;
    v16h a;
#pragma unroll
    for (int e = 0; e < 16; ++e) {
        const int kk = e;
        a[e] = (_Float16)row[kk];
    }
    return a;
}

__device__ inline v16h load_b_frag(const float* __restrict__ W, int n0, int kc, int lane) {
    const int half = lane >> 4;
    const int n    = n0 + (lane & 15);
    const float* wrow = W + (size_t)n * CDIM + kc * 32 + half * 16;
    v16h b;
#pragma unroll
    for (int e = 0; e < 16; ++e) b[e] = (_Float16)wrow[e];
    return b;
}

// ---------------------------------------------------------------------------
// Kernel 1: fused Q/K/V projection.  Y{q,k,v}[m, n] = sum_c feats[m,c]*W{q,k,v}[n,c]
// Block = 256 threads = 8 waves. Block owns rows [16*bx, 16*bx+16).
// Wave w owns output columns [16w, 16w+16) for Q, K and V (A frag reused 3x).
// ---------------------------------------------------------------------------
__global__ __launch_bounds__(256) void qkv_wmma_kernel(
    const float* __restrict__ feats,
    const float* __restrict__ Wq, const float* __restrict__ Wk, const float* __restrict__ Wv,
    float* __restrict__ Q, float* __restrict__ K, float* __restrict__ V) {

    const int wave = threadIdx.x >> 5;
    const int lane = threadIdx.x & 31;
    const int m0   = blockIdx.x * 16;
    const int n0   = wave * 16;

    v8f accq = {}; v8f acck = {}; v8f accv = {};
#pragma unroll
    for (int kc = 0; kc < 4; ++kc) {
        v16h a  = load_a_frag(feats, m0, kc, lane);
        v16h bq = load_b_frag(Wq,   n0, kc, lane);
        v16h bk = load_b_frag(Wk,   n0, kc, lane);
        v16h bv = load_b_frag(Wv,   n0, kc, lane);
        accq = __builtin_amdgcn_wmma_f32_16x16x32_f16(false, a, false, bq, (short)0, accq, false, false);
        acck = __builtin_amdgcn_wmma_f32_16x16x32_f16(false, a, false, bk, (short)0, acck, false, false);
        accv = __builtin_amdgcn_wmma_f32_16x16x32_f16(false, a, false, bv, (short)0, accv, false, false);
    }

    const int half = lane >> 4;
    const int n    = n0 + (lane & 15);
#pragma unroll
    for (int r = 0; r < 8; ++r) {
        const size_t off = (size_t)(m0 + r + half * 8) * CDIM + n;
        Q[off] = accq[r];
        K[off] = acck[r];
        V[off] = accv[r];
    }
}

// ---------------------------------------------------------------------------
// Kernel 2: gather-attention. One wave per node; 8 nodes per 256-thread block.
// scores[h][k] = (q[n,h,:] . K[knn[n][k],h,:]) / sqrt(D); softmax over k;
// out[n,h,d] = sum_k w[h][k] * V[knn[n][k], h*32+d]
// ---------------------------------------------------------------------------
__global__ __launch_bounds__(256) void attn_kernel(
    const float* __restrict__ Q, const float* __restrict__ Kf, const float* __restrict__ Vf,
    const long long* __restrict__ knn, float* __restrict__ out) {

    __shared__ float sw[8][HEADS][KNN];   // per-wave score/weight scratch (2 KB)

    const int wave = threadIdx.x >> 5;
    const int lane = threadIdx.x & 31;
    const int node = blockIdx.x * 8 + wave;

    const float* qrow = Q + (size_t)node * CDIM;
    const long long* nidx = knn + (size_t)node * KNN;

    // Phase 1: 64 (h,k) scores, 2 per lane
#pragma unroll
    for (int p = 0; p < 2; ++p) {
        const int idx = p * 32 + lane;        // 0..63
        const int h = idx >> 4;
        const int k = idx & 15;
        const long long nb = nidx[k];
        const float* krow = Kf + (size_t)nb * CDIM + h * DHEAD;
        const float* qh   = qrow + h * DHEAD;
        float s = 0.f;
#pragma unroll
        for (int d = 0; d < DHEAD; ++d) s = fmaf(qh[d], krow[d], s);
        sw[wave][h][k] = s * 0.17677669529663687f;   // 1/sqrt(32)
    }
    __syncthreads();

    // Phase 2: softmax over K neighbors; lanes 0..3 each own one head
    if (lane < HEADS) {
        float mx = -INFINITY;
#pragma unroll
        for (int k = 0; k < KNN; ++k) mx = fmaxf(mx, sw[wave][lane][k]);
        float e[KNN];
        float sum = 0.f;
#pragma unroll
        for (int k = 0; k < KNN; ++k) { e[k] = __expf(sw[wave][lane][k] - mx); sum += e[k]; }
        const float inv = 1.f / sum;
#pragma unroll
        for (int k = 0; k < KNN; ++k) sw[wave][lane][k] = e[k] * inv;
    }
    __syncthreads();

    // Phase 3: weighted V sum. lane = d within head; loop over heads.
    float* orow = out + (size_t)node * CDIM;
#pragma unroll
    for (int h = 0; h < HEADS; ++h) {
        float acc = 0.f;
#pragma unroll
        for (int k = 0; k < KNN; ++k) {
            const long long nb = nidx[k];
            acc = fmaf(sw[wave][h][k], Vf[(size_t)nb * CDIM + h * DHEAD + lane], acc);
        }
        orow[h * DHEAD + lane] = acc;
    }
}

// ---------------------------------------------------------------------------
// Kernel 3: output projection with bias. Y[m,n] = sum_c X[m,c]*Wo[n,c] + bo[n]
// ---------------------------------------------------------------------------
__global__ __launch_bounds__(256) void oproj_wmma_kernel(
    const float* __restrict__ X, const float* __restrict__ Wo,
    const float* __restrict__ bo, float* __restrict__ Y) {

    const int wave = threadIdx.x >> 5;
    const int lane = threadIdx.x & 31;
    const int m0   = blockIdx.x * 16;
    const int n0   = wave * 16;

    v8f acc = {};
#pragma unroll
    for (int kc = 0; kc < 4; ++kc) {
        v16h a = load_a_frag(X,  m0, kc, lane);
        v16h b = load_b_frag(Wo, n0, kc, lane);
        acc = __builtin_amdgcn_wmma_f32_16x16x32_f16(false, a, false, b, (short)0, acc, false, false);
    }

    const int half = lane >> 4;
    const int n    = n0 + (lane & 15);
    const float bias = bo[n];
#pragma unroll
    for (int r = 0; r < 8; ++r) {
        const size_t off = (size_t)(m0 + r + half * 8) * CDIM + n;
        Y[off] = acc[r] + bias;
    }
}

// ---------------------------------------------------------------------------
extern "C" void kernel_launch(void* const* d_in, const int* in_sizes, int n_in,
                              void* d_out, int out_size, void* d_ws, size_t ws_size,
                              hipStream_t stream) {
    // setup_inputs order: feats, coords, knn_idx, Wq, Wk, Wv, Wo, bo
    const float*     feats = (const float*)d_in[0];
    // d_in[1] = coords: only feeds the precomputed kNN; unused here
    const long long* knn   = (const long long*)d_in[2];   // int64 indices
    const float*     Wq    = (const float*)d_in[3];
    const float*     Wk    = (const float*)d_in[4];
    const float*     Wv    = (const float*)d_in[5];
    const float*     Wo    = (const float*)d_in[6];
    const float*     bo    = (const float*)d_in[7];
    float*           out   = (float*)d_out;

    const size_t NC = (size_t)NPTS * CDIM;   // 4,194,304 floats = 16 MiB
    float* Q   = (float*)d_ws;
    float* K   = Q + NC;
    float* V   = K + NC;
    float* att = V + NC;

    dim3 blk(256);
    // 1) fused QKV projection: 2048 row-tiles of 16
    qkv_wmma_kernel<<<dim3(NPTS / 16), blk, 0, stream>>>(feats, Wq, Wk, Wv, Q, K, V);
    // 2) gather attention: one wave per node, 8 nodes per block
    attn_kernel<<<dim3(NPTS / 8), blk, 0, stream>>>(Q, K, V, knn, att);
    // 3) output projection + bias
    oproj_wmma_kernel<<<dim3(NPTS / 16), blk, 0, stream>>>(att, Wo, bo, out);
}
